// BertCRF_61031485276762
// MI455X (gfx1250) — matrix operand order repaired and checked by reading, not compiled, not run on hardware
//
#include <hip/hip_runtime.h>
#include <hip/hip_bf16.h>
#include <stdint.h>

// ---------------- problem constants ----------------
#define B_   32
#define S_   256
#define L_   12
#define H_   768
#define NH_  12
#define DH_  64
#define FF_  3072
#define T_   9
#define MTOK (B_*S_)          // 8192 token rows

// ---------------- CDNA5 WMMA plumbing ----------------
typedef __attribute__((ext_vector_type(16))) __bf16 v16bf;
typedef __attribute__((ext_vector_type(8)))  float  v8f;
typedef __attribute__((ext_vector_type(4)))  int    v4i_;

union FragBF { v16bf v; uint4 q[2]; };

// ---- async global->LDS (ASYNCcnt path), guarded so the build can't regress ----
#if defined(__has_builtin)
#  if __has_builtin(__builtin_amdgcn_global_load_async_to_lds_b128)
#    define ASYNC_LDS 1
#  endif
#endif
#ifndef ASYNC_LDS
#  define ASYNC_LDS 0
#endif

__device__ __forceinline__ void cp_async_b128(void* lds_dst, const void* g_src) {
#if ASYNC_LDS
  __builtin_amdgcn_global_load_async_to_lds_b128(
      (__attribute__((address_space(1))) v4i_*)g_src,
      (__attribute__((address_space(3))) v4i_*)lds_dst,
      /*imm offset*/0, /*cpol*/0);
#else
  *(uint4*)lds_dst = *(const uint4*)g_src;
#endif
}

__device__ __forceinline__ void cp_async_wait() {
#if ASYNC_LDS
#  if __has_builtin(__builtin_amdgcn_s_wait_asynccnt)
  __builtin_amdgcn_s_wait_asynccnt(0);
#  else
  asm volatile("s_wait_asynccnt 0x0" ::: "memory");
#  endif
#endif
}

__device__ __forceinline__ unsigned short f2bf(float f) {
  union { float f; unsigned u; } x; x.f = f;
  unsigned r = x.u + 0x7fffu + ((x.u >> 16) & 1u);   // round-to-nearest-even
  return (unsigned short)(r >> 16);
}

__device__ __forceinline__ v8f wmma_bf(v16bf a, v16bf b, v8f c) {
  // D = A(16x32 bf16) * B(32x16 bf16) + C(16x16 f32)
  return __builtin_amdgcn_wmma_f32_16x16x32_bf16(false, a, false, b,
                                                 (short)0, c, false, false);
}

// A fragment (16x32, bf16) from row-major LDS tile (ldm elements per row).
// Lane layout (ISA 7.12.2): lanes 0-15 row=lane, K={kh..kh+7, 16+kh..16+kh+7}, kh=8*(lane>=16)
__device__ __forceinline__ v16bf load_frag_a(const unsigned short* lds, int ldm, int lane) {
  int row = lane & 15, kh = (lane >> 4) << 3;
  FragBF fr;
  fr.q[0] = *(const uint4*)(lds + row * ldm + kh);
  fr.q[1] = *(const uint4*)(lds + row * ldm + 16 + kh);
  return fr.v;
}

// Same as load_frag_a but source is f32 in LDS (converted on the fly).
__device__ __forceinline__ v16bf load_frag_a_f32(const float* lds, int ldm, int lane) {
  int row = lane & 15, kh = (lane >> 4) << 3;
  FragBF fr;
#pragma unroll
  for (int j = 0; j < 8; ++j) fr.v[j]     = (__bf16)lds[row * ldm + kh + j];
#pragma unroll
  for (int j = 0; j < 8; ++j) fr.v[8 + j] = (__bf16)lds[row * ldm + 16 + kh + j];
  return fr.v;
}

// B fragment (32x16, bf16) from a K-contiguous (transposed, [n][k]) LDS tile.
// Lane layout: col = lane&15 (+tile base), K={kb..kb+7, 16+kb..16+kb+7}, kb=8*(lane>=16)
__device__ __forceinline__ v16bf load_frag_b(const unsigned short* lds, int ldk, int lane) {
  int col = lane & 15, kb = (lane >> 4) << 3;
  FragBF fr;
  fr.q[0] = *(const uint4*)(lds + col * ldk + kb);
  fr.q[1] = *(const uint4*)(lds + col * ldk + kb + 16);
  return fr.v;
}

// ---------------- embedding + LayerNorm ----------------
__global__ __launch_bounds__(256) void embed_ln_kernel(
    const int* __restrict__ input_x,
    const float* __restrict__ wemb, const float* __restrict__ pemb,
    const float* __restrict__ temb,
    const float* __restrict__ g, const float* __restrict__ bta,
    float* __restrict__ x, unsigned short* __restrict__ xh) {
  int row = blockIdx.x;                 // token index 0..8191
  int tid = threadIdx.x;
  int s   = row & (S_ - 1);
  int tok = input_x[row];
  __shared__ float red[256];
  float v[3]; float sum = 0.f;
#pragma unroll
  for (int j = 0; j < 3; ++j) {
    int c = tid + j * 256;
    v[j] = wemb[(size_t)tok * H_ + c] + pemb[(size_t)s * H_ + c] + temb[c];
    sum += v[j];
  }
  red[tid] = sum; __syncthreads();
  for (int o = 128; o > 0; o >>= 1) { if (tid < o) red[tid] += red[tid + o]; __syncthreads(); }
  float mean = red[0] * (1.f / H_); __syncthreads();
  float q = 0.f;
#pragma unroll
  for (int j = 0; j < 3; ++j) { float d = v[j] - mean; q += d * d; }
  red[tid] = q; __syncthreads();
  for (int o = 128; o > 0; o >>= 1) { if (tid < o) red[tid] += red[tid + o]; __syncthreads(); }
  float inv = rsqrtf(red[0] * (1.f / H_) + 1e-12f);
#pragma unroll
  for (int j = 0; j < 3; ++j) {
    int c = tid + j * 256;
    float o = (v[j] - mean) * inv * g[c] + bta[c];
    x[(size_t)row * H_ + c]  = o;
    xh[(size_t)row * H_ + c] = f2bf(o);
  }
}

// ---------------- residual + LayerNorm ----------------
__global__ __launch_bounds__(256) void ln_res_kernel(
    const float* __restrict__ res, const float* __restrict__ delta,
    const float* __restrict__ g, const float* __restrict__ bta,
    float* __restrict__ xo, unsigned short* __restrict__ xh) {
  int row = blockIdx.x, tid = threadIdx.x;
  __shared__ float red[256];
  float v[3]; float sum = 0.f;
#pragma unroll
  for (int j = 0; j < 3; ++j) {
    int c = tid + j * 256;
    v[j] = res[(size_t)row * H_ + c] + delta[(size_t)row * H_ + c];
    sum += v[j];
  }
  red[tid] = sum; __syncthreads();
  for (int o = 128; o > 0; o >>= 1) { if (tid < o) red[tid] += red[tid + o]; __syncthreads(); }
  float mean = red[0] * (1.f / H_); __syncthreads();
  float q = 0.f;
#pragma unroll
  for (int j = 0; j < 3; ++j) { float d = v[j] - mean; q += d * d; }
  red[tid] = q; __syncthreads();
  for (int o = 128; o > 0; o >>= 1) { if (tid < o) red[tid] += red[tid + o]; __syncthreads(); }
  float inv = rsqrtf(red[0] * (1.f / H_) + 1e-12f);
#pragma unroll
  for (int j = 0; j < 3; ++j) {
    int c = tid + j * 256;
    float o = (v[j] - mean) * inv * g[c] + bta[c];
    xo[(size_t)row * H_ + c]  = o;
    xh[(size_t)row * H_ + c] = f2bf(o);
  }
}

// ---------------- WMMA GEMM: C[M,N] = A_bf16[M,K] * W_f32[K,N] + bias ----------------
// MODE 0: f32 out ; MODE 1: bf16 out ; MODE 2: gelu -> bf16 out
template <int MODE>
__global__ __launch_bounds__(256) void gemm128_kernel(
    const unsigned short* __restrict__ A, const float* __restrict__ W,
    const float* __restrict__ bias,
    float* __restrict__ outF, unsigned short* __restrict__ outH,
    int M, int N, int K) {
  __shared__ unsigned short sA[128 * 32];   // [m][k]
  __shared__ unsigned short sB[128 * 32];   // transposed [n][k]
  int m0 = blockIdx.x * 128, n0 = blockIdx.y * 128;
  int tid = threadIdx.x, lane = tid & 31, wave = tid >> 5;
  int wm = wave & 3, wn = wave >> 2;       // 4 x 2 wave grid -> 32x64 per wave

  v8f acc[2][4];
#pragma unroll
  for (int i = 0; i < 2; ++i)
#pragma unroll
    for (int j = 0; j < 4; ++j) acc[i][j] = (v8f)0.f;

  int arow = tid >> 1, aseg = (tid & 1) * 16;        // A stage mapping
  int wk = tid >> 3, wns = (tid & 7) * 16;           // W stage mapping

  int ksteps = K >> 5;
  for (int kt = 0; kt < ksteps; ++kt) {
    int k0 = kt << 5;
    // prefetch next K-slab of the weight matrix (global_prefetch_b8)
    if (kt + 1 < ksteps) {
      __builtin_prefetch(W + (size_t)(k0 + 32 + wk) * N + n0 + wns, 0, 1);
    }
    __syncthreads();
    {   // stage A tile 128x32 (bf16, row-major) — async direct-to-LDS
      const unsigned short* src = A + (size_t)(m0 + arow) * K + k0 + aseg;
      unsigned short* dst = sA + arow * 32 + aseg;
      cp_async_b128(dst,     src);
      cp_async_b128(dst + 8, src + 8);
    }
    {   // stage W tile 32x128 (f32) -> bf16, transposed [n][k]
      const float* src = W + (size_t)(k0 + wk) * N + n0 + wns;
#pragma unroll
      for (int j = 0; j < 16; ++j) sB[(wns + j) * 32 + wk] = f2bf(src[j]);
    }
    cp_async_wait();
    __syncthreads();
    v16bf a0 = load_frag_a(sA + (wm * 32 + 0)  * 32, 32, lane);
    v16bf a1 = load_frag_a(sA + (wm * 32 + 16) * 32, 32, lane);
    v16bf bfr[4];
#pragma unroll
    for (int j = 0; j < 4; ++j)
      bfr[j] = load_frag_b(sB + (wn * 64 + j * 16) * 32, 32, lane);
#pragma unroll
    for (int j = 0; j < 4; ++j) {
      acc[0][j] = wmma_bf(a0, bfr[j], acc[0][j]);
      acc[1][j] = wmma_bf(a1, bfr[j], acc[1][j]);
    }
  }
  // epilogue
  int col = lane & 15, rh = (lane >> 4) << 3;
#pragma unroll
  for (int i = 0; i < 2; ++i)
#pragma unroll
    for (int j = 0; j < 4; ++j) {
      int gc = n0 + wn * 64 + j * 16 + col;
      float bj = bias[gc];
#pragma unroll
      for (int e = 0; e < 8; ++e) {
        int gr = m0 + wm * 32 + i * 16 + e + rh;
        float v = acc[i][j][e] + bj;
        if (MODE == 2) v = 0.5f * v * (1.f + erff(v * 0.70710678118f));
        if (MODE == 0) outF[(size_t)gr * N + gc] = v;
        else           outH[(size_t)gr * N + gc] = f2bf(v);
      }
    }
}

// ---------------- fused attention (per b, head, 64-row q-block) ----------------
__global__ __launch_bounds__(256) void attn_kernel(
    const unsigned short* __restrict__ qg, const unsigned short* __restrict__ kg,
    const unsigned short* __restrict__ vg, const int* __restrict__ input_x,
    unsigned short* __restrict__ ctx) {
  extern __shared__ unsigned char smem[];
  unsigned short* sQ  = (unsigned short*)smem;      // [64][64]
  unsigned short* sK  = sQ + 64 * 64;               // [64][64]
  unsigned short* sVt = sK + 64 * 64;               // [64 d][256 key]
  float* sS    = (float*)(sVt + 64 * 256);          // [64][256] scores
  float* sRed  = sS + 64 * 256;                     // [64][4]
  float* sStat = sRed + 64 * 4;                     // [64]

  int bidx = blockIdx.x;
  int qb = bidx & 3, h = (bidx >> 2) % NH_, b = bidx / (4 * NH_);
  int tid = threadIdx.x, lane = tid & 31, wave = tid >> 5;
  int hoff = h * DH_;
  int qrow0 = qb * 64;
  int srow = tid >> 2, sseg = (tid & 3) * 16;       // staging mapping

  {   // stage Q tile (64x64 bf16) — async direct-to-LDS
    const unsigned short* src = qg + (size_t)(b * S_ + qrow0 + srow) * H_ + hoff + sseg;
    unsigned short* dst = sQ + srow * 64 + sseg;
    cp_async_b128(dst,     src);
    cp_async_b128(dst + 8, src + 8);
  }

  // phase 1: scores = Q K^T / 8 + mask bias
  for (int kt = 0; kt < 4; ++kt) {
    __syncthreads();
    {
      const unsigned short* src = kg + (size_t)(b * S_ + kt * 64 + srow) * H_ + hoff + sseg;
      unsigned short* dst = sK + srow * 64 + sseg;
      cp_async_b128(dst,     src);
      cp_async_b128(dst + 8, src + 8);
      const unsigned short* vs = vg + (size_t)(b * S_ + kt * 64 + srow) * H_ + hoff + sseg;
#pragma unroll
      for (int j = 0; j < 16; ++j) sVt[(sseg + j) * 256 + kt * 64 + srow] = vs[j];
    }
    cp_async_wait();
    __syncthreads();
#pragma unroll
    for (int tI = 0; tI < 2; ++tI) {
      int ti = wave * 2 + tI, qi = ti >> 2, ki = ti & 3;
      v8f acc = (v8f)0.f;
#pragma unroll
      for (int ks = 0; ks < 2; ++ks) {
        v16bf a  = load_frag_a(sQ + qi * 16 * 64 + ks * 32, 64, lane);
        v16bf bb = load_frag_b(sK + ki * 16 * 64 + ks * 32, 64, lane);
        acc = wmma_bf(a, bb, acc);
      }
      int col = lane & 15, rh = (lane >> 4) << 3;
#pragma unroll
      for (int e = 0; e < 8; ++e) {
        int r   = qi * 16 + e + rh;
        int kgl = kt * 64 + ki * 16 + col;
        float mb = (input_x[b * S_ + kgl] != 0) ? 0.f : -1e9f;
        sS[r * 256 + kgl] = acc[e] * 0.125f + mb;
      }
    }
  }
  __syncthreads();

  // softmax over 256 keys, 4 threads per row
  {
    int r = tid >> 2, seg = tid & 3;
    float m = -1e30f;
    for (int j = 0; j < 64; ++j) m = fmaxf(m, sS[r * 256 + seg * 64 + j]);
    sRed[r * 4 + seg] = m;
  }
  __syncthreads();
  if (tid < 64)
    sStat[tid] = fmaxf(fmaxf(sRed[tid * 4], sRed[tid * 4 + 1]),
                       fmaxf(sRed[tid * 4 + 2], sRed[tid * 4 + 3]));
  __syncthreads();
  {
    int r = tid >> 2, seg = tid & 3;
    float m = sStat[r], sum = 0.f;
    for (int j = 0; j < 64; ++j) {
      float e = __expf(sS[r * 256 + seg * 64 + j] - m);
      sS[r * 256 + seg * 64 + j] = e; sum += e;
    }
    sRed[r * 4 + seg] = sum;
  }
  __syncthreads();
  if (tid < 64)
    sStat[tid] = 1.f / (sRed[tid * 4] + sRed[tid * 4 + 1] + sRed[tid * 4 + 2] + sRed[tid * 4 + 3]);
  __syncthreads();
  {
    int r = tid >> 2, seg = tid & 3; float inv = sStat[r];
    for (int j = 0; j < 64; ++j) sS[r * 256 + seg * 64 + j] *= inv;
  }
  __syncthreads();

  // phase 2: ctx = P V  (K-dim 256, A = f32 scores -> bf16 frags, B = V^T)
#pragma unroll
  for (int tI = 0; tI < 2; ++tI) {
    int ti = wave * 2 + tI, qi = ti >> 2, di = ti & 3;
    v8f acc = (v8f)0.f;
#pragma unroll
    for (int ks = 0; ks < 8; ++ks) {
      v16bf a  = load_frag_a_f32(sS + qi * 16 * 256 + ks * 32, 256, lane);
      v16bf bb = load_frag_b(sVt + di * 16 * 256 + ks * 32, 256, lane);
      acc = wmma_bf(a, bb, acc);
    }
    int col = lane & 15, rh = (lane >> 4) << 3;
#pragma unroll
    for (int e = 0; e < 8; ++e) {
      int r = qi * 16 + e + rh;
      ctx[(size_t)(b * S_ + qrow0 + r) * H_ + hoff + di * 16 + col] = f2bf(acc[e]);
    }
  }
}

// ---------------- emissions (N = 9, tiny) ----------------
__global__ __launch_bounds__(256) void emit_kernel(
    const float* __restrict__ x, const float* __restrict__ w,
    const float* __restrict__ b, float* __restrict__ em) {
  int row = blockIdx.x, tid = threadIdx.x;
  __shared__ float sx[H_];
#pragma unroll
  for (int j = 0; j < 3; ++j) sx[tid + j * 256] = x[(size_t)row * H_ + tid + j * 256];
  __syncthreads();
  if (tid < T_) {
    float acc = b[tid];
    for (int c = 0; c < H_; ++c) acc += sx[c] * w[c * T_ + tid];
    em[row * T_ + tid] = acc;
  }
}

// ---------------- CRF negative log-likelihood ----------------
__global__ __launch_bounds__(512) void crf_loss_kernel(
    const float* __restrict__ em, const int* __restrict__ label,
    const int* __restrict__ input_x,
    const float* __restrict__ start, const float* __restrict__ endv,
    const float* __restrict__ trans, float* __restrict__ out) {
  __shared__ float alpha[B_][12];
  __shared__ float nxt[B_][12];
  __shared__ float bl[B_];
  int tid = threadIdx.x;
  int b = tid >> 4, t = tid & 15;
  bool act = (b < B_) && (t < T_);
  if (act) alpha[b][t] = start[t] + em[(size_t)(b * S_) * T_ + t];
  __syncthreads();
  for (int s = 1; s < S_; ++s) {
    if (act) {
      float m = -1e30f;
#pragma unroll
      for (int i = 0; i < T_; ++i) m = fmaxf(m, alpha[b][i] + trans[i * T_ + t]);
      float acc = 0.f;
#pragma unroll
      for (int i = 0; i < T_; ++i) acc += __expf(alpha[b][i] + trans[i * T_ + t] - m);
      float v = m + __logf(acc) + em[(size_t)(b * S_ + s) * T_ + t];
      nxt[b][t] = (input_x[b * S_ + s] != 0) ? v : alpha[b][t];
    }
    __syncthreads();
    if (act) alpha[b][t] = nxt[b][t];
    __syncthreads();
  }
  if (act && t == 0) {
    float m = -1e30f;
#pragma unroll
    for (int i = 0; i < T_; ++i) m = fmaxf(m, alpha[b][i] + endv[i]);
    float acc = 0.f;
#pragma unroll
    for (int i = 0; i < T_; ++i) acc += __expf(alpha[b][i] + endv[i] - m);
    float denom = m + __logf(acc);
    int lp = label[b * S_];
    float num = start[lp] + em[(size_t)(b * S_) * T_ + lp];
    int last = lp;
    for (int s = 1; s < S_; ++s) {
      int ls = label[b * S_ + s];
      if (input_x[b * S_ + s] != 0) {
        num += trans[lp * T_ + ls] + em[(size_t)(b * S_ + s) * T_ + ls];
        last = ls;
      }
      lp = ls;
    }
    num += endv[last];
    bl[b] = denom - num;
  }
  __syncthreads();
  if (tid == 0) {
    float tot = 0.f;
    for (int i = 0; i < B_; ++i) tot += bl[i];
    out[0] = tot;
  }
}

// ---------------- Viterbi decode ----------------
__global__ __launch_bounds__(32) void viterbi_kernel(
    const float* __restrict__ em,
    const float* __restrict__ start, const float* __restrict__ endv,
    const float* __restrict__ trans, float* __restrict__ out) {
  int b = blockIdx.x, t = threadIdx.x;
  __shared__ float sc[16], ns[16];
  __shared__ unsigned char bp[S_][16];
  __shared__ int path[S_];
  if (t < T_) sc[t] = start[t] + em[(size_t)(b * S_) * T_ + t];
  __syncthreads();
  for (int s = 1; s < S_; ++s) {
    if (t < T_) {
      float best = -1e30f; int bi = 0;
#pragma unroll
      for (int i = 0; i < T_; ++i) {
        float v = sc[i] + trans[i * T_ + t];
        if (v > best) { best = v; bi = i; }
      }
      ns[t] = best + em[(size_t)(b * S_ + s) * T_ + t];
      bp[s][t] = (unsigned char)bi;
    }
    __syncthreads();
    if (t < T_) sc[t] = ns[t];
    __syncthreads();
  }
  if (t == 0) {
    float best = -1e30f; int bi = 0;
#pragma unroll
    for (int i = 0; i < T_; ++i) {
      float v = sc[i] + endv[i];
      if (v > best) { best = v; bi = i; }
    }
    int cur = bi; path[S_ - 1] = cur;
    for (int s = S_ - 1; s >= 1; --s) { cur = bp[s][cur]; path[s - 1] = cur; }
    for (int s = 0; s < S_; ++s) out[1 + b * S_ + s] = (float)path[s];
  }
}

// ---------------- host orchestration ----------------
extern "C" void kernel_launch(void* const* d_in, const int* in_sizes, int n_in,
                              void* d_out, int out_size, void* d_ws, size_t ws_size,
                              hipStream_t stream) {
  (void)in_sizes; (void)n_in; (void)out_size; (void)ws_size;
  const int*   input_x = (const int*)d_in[0];
  const int*   label   = (const int*)d_in[1];
  const float* wemb    = (const float*)d_in[2];
  const float* pemb    = (const float*)d_in[3];
  const float* temb    = (const float*)d_in[4];
  const float* eln_g   = (const float*)d_in[5];
  const float* eln_b   = (const float*)d_in[6];
  const float* wq = (const float*)d_in[7];   const float* bq = (const float*)d_in[8];
  const float* wk = (const float*)d_in[9];   const float* bk = (const float*)d_in[10];
  const float* wv = (const float*)d_in[11];  const float* bv = (const float*)d_in[12];
  const float* wo = (const float*)d_in[13];  const float* bo = (const float*)d_in[14];
  const float* l1g = (const float*)d_in[15]; const float* l1b = (const float*)d_in[16];
  const float* w1 = (const float*)d_in[17];  const float* b1 = (const float*)d_in[18];
  const float* w2 = (const float*)d_in[19];  const float* b2 = (const float*)d_in[20];
  const float* l2g = (const float*)d_in[21]; const float* l2b = (const float*)d_in[22];
  const float* emit_w = (const float*)d_in[23];
  const float* emit_b = (const float*)d_in[24];
  const float* crf_start = (const float*)d_in[25];
  const float* crf_end   = (const float*)d_in[26];
  const float* crf_trans = (const float*)d_in[27];
  float* out = (float*)d_out;

  // workspace carve-out
  char* base = (char*)d_ws; size_t off = 0;
  auto alloc = [&](size_t bytes) -> void* {
    void* p = base + off;
    off = (off + bytes + 255) & ~(size_t)255;
    return p;
  };
  float*          x    = (float*)alloc((size_t)MTOK * H_ * 4);
  unsigned short* xh   = (unsigned short*)alloc((size_t)MTOK * H_ * 2);
  unsigned short* qh   = (unsigned short*)alloc((size_t)MTOK * H_ * 2);
  unsigned short* kh   = (unsigned short*)alloc((size_t)MTOK * H_ * 2);
  unsigned short* vh   = (unsigned short*)alloc((size_t)MTOK * H_ * 2);
  unsigned short* ctxh = (unsigned short*)alloc((size_t)MTOK * H_ * 2);
  float*          tmp  = (float*)alloc((size_t)MTOK * H_ * 4);
  unsigned short* ffh  = (unsigned short*)alloc((size_t)MTOK * FF_ * 2);
  float*          em   = (float*)alloc((size_t)MTOK * T_ * 4);

  size_t att_lds = (size_t)(64 * 64 + 64 * 64 + 64 * 256) * 2 +
                   (size_t)(64 * 256 + 64 * 4 + 64) * 4;

  embed_ln_kernel<<<MTOK, 256, 0, stream>>>(input_x, wemb, pemb, temb, eln_g, eln_b, x, xh);

  dim3 gHH(MTOK / 128, H_ / 128);    // 64 x 6
  dim3 gHF(MTOK / 128, FF_ / 128);   // 64 x 24
  for (int l = 0; l < L_; ++l) {
    const float* wq_l = wq + (size_t)l * H_ * H_;  const float* bq_l = bq + (size_t)l * H_;
    const float* wk_l = wk + (size_t)l * H_ * H_;  const float* bk_l = bk + (size_t)l * H_;
    const float* wv_l = wv + (size_t)l * H_ * H_;  const float* bv_l = bv + (size_t)l * H_;
    const float* wo_l = wo + (size_t)l * H_ * H_;  const float* bo_l = bo + (size_t)l * H_;
    const float* w1_l = w1 + (size_t)l * H_ * FF_; const float* b1_l = b1 + (size_t)l * FF_;
    const float* w2_l = w2 + (size_t)l * FF_ * H_; const float* b2_l = b2 + (size_t)l * H_;
    const float* l1g_l = l1g + (size_t)l * H_;     const float* l1b_l = l1b + (size_t)l * H_;
    const float* l2g_l = l2g + (size_t)l * H_;     const float* l2b_l = l2b + (size_t)l * H_;

    gemm128_kernel<1><<<gHH, 256, 0, stream>>>(xh, wq_l, bq_l, nullptr, qh, MTOK, H_, H_);
    gemm128_kernel<1><<<gHH, 256, 0, stream>>>(xh, wk_l, bk_l, nullptr, kh, MTOK, H_, H_);
    gemm128_kernel<1><<<gHH, 256, 0, stream>>>(xh, wv_l, bv_l, nullptr, vh, MTOK, H_, H_);

    attn_kernel<<<B_ * NH_ * (S_ / 64), 256, att_lds, stream>>>(qh, kh, vh, input_x, ctxh);

    gemm128_kernel<0><<<gHH, 256, 0, stream>>>(ctxh, wo_l, bo_l, tmp, nullptr, MTOK, H_, H_);
    ln_res_kernel<<<MTOK, 256, 0, stream>>>(x, tmp, l1g_l, l1b_l, x, xh);

    gemm128_kernel<2><<<gHF, 256, 0, stream>>>(xh, w1_l, b1_l, nullptr, ffh, MTOK, FF_, H_);
    gemm128_kernel<0><<<gHH, 256, 0, stream>>>(ffh, w2_l, b2_l, tmp, nullptr, MTOK, H_, FF_);
    ln_res_kernel<<<MTOK, 256, 0, stream>>>(x, tmp, l2g_l, l2b_l, x, xh);
  }

  emit_kernel<<<MTOK, 256, 0, stream>>>(x, emit_w, emit_b, em);
  crf_loss_kernel<<<1, 512, 0, stream>>>(em, label, input_x, crf_start, crf_end, crf_trans, out);
  viterbi_kernel<<<B_, 32, 0, stream>>>(em, crf_start, crf_end, crf_trans, out);
}
